// OptimizedDGConvModule_3375844294918
// MI455X (gfx1250) — matrix operand order repaired and checked by reading, not compiled, not run on hardware
//
#include <hip/hip_runtime.h>
#include <hip/hip_bf16.h>
#include <math.h>

typedef __attribute__((ext_vector_type(16))) __bf16 v16bf;
typedef __attribute__((ext_vector_type(8)))  __bf16 v8bf;
typedef __attribute__((ext_vector_type(8)))  float  v8f;

#define CAT16(lo, hi) __builtin_shufflevector(lo, hi, 0,1,2,3,4,5,6,7,8,9,10,11,12,13,14,15)

// float -> bf16 bits, round-to-nearest-even
__device__ __forceinline__ unsigned short f2bf(float f) {
    unsigned u = __float_as_uint(f);
    u = (u + 0x7FFFu + ((u >> 16) & 1u)) >> 16;
    return (unsigned short)u;
}

// ---------------------------------------------------------------------------
// Kernel A: per-(b,c) global average pool; also convert fus_w to bf16 once.
// grid = 512 (= B*C), block = 256
// ---------------------------------------------------------------------------
__global__ __launch_bounds__(256) void dgconv_pool_kernel(
    const float* __restrict__ x, const float* __restrict__ fus_w,
    float* __restrict__ pooled, unsigned short* __restrict__ fusw_bf) {
    const int plane = blockIdx.x;          // b*64 + c
    const int t = threadIdx.x;
    const float4* p4 = (const float4*)(x + (size_t)plane * 65536u);
    float s = 0.f;
    #pragma unroll 8
    for (int i = 0; i < 64; ++i) {         // 64 * 256 float4 = 65536 floats
        float4 v = p4[i * 256 + t];
        s += (v.x + v.y) + (v.z + v.w);
    }
    __shared__ float red[256];
    red[t] = s;
    __syncthreads();
    for (int off = 128; off > 0; off >>= 1) {
        if (t < off) red[t] += red[t + off];
        __syncthreads();
    }
    if (t == 0) pooled[plane] = red[0] * (1.0f / 65536.0f);

    const int gid = blockIdx.x * 256 + t;  // fus_w is 64*64 = 4096 elements
    if (gid < 4096) fusw_bf[gid] = f2bf(fus_w[gid]);
}

// ---------------------------------------------------------------------------
// Kernel B: SE attention MLP + combined per-(b,c) 3x3 kernels.
// grid = B (8), block = 64 (one thread per channel)
// kcomb layout: [B][64][12] floats (9 used, stride 12 -> 48B rows, 16B aligned)
// ---------------------------------------------------------------------------
__global__ __launch_bounds__(64) void dgconv_att_kernel(
    const float* __restrict__ pooled,
    const float* __restrict__ idg_w, const float* __restrict__ van_w,
    const float* __restrict__ w1, const float* __restrict__ b1,
    const float* __restrict__ w2, const float* __restrict__ b2,
    float* __restrict__ kcomb) {
    const int b = blockIdx.x;
    const int t = threadIdx.x;             // channel
    __shared__ float sp[64];
    __shared__ float sh[16];
    sp[t] = pooled[b * 64 + t];
    __syncthreads();
    if (t < 16) {
        float a = b1[t];
        #pragma unroll 8
        for (int c = 0; c < 64; ++c) a = fmaf(w1[t * 64 + c], sp[c], a);
        sh[t] = fmaxf(a, 0.0f);
    }
    __syncthreads();
    float att[6];
    #pragma unroll
    for (int f = 0; f < 6; ++f) {
        const int o = f * 64 + t;
        float a = b2[o];
        #pragma unroll
        for (int j = 0; j < 16; ++j) a = fmaf(w2[o * 16 + j], sh[j], a);
        att[f] = 1.0f / (1.0f + expf(-a));
    }
    // fixed stencils (same normalization as reference)
    const float SX[9] = {-0.125f, 0.f, 0.125f, -0.25f, 0.f, 0.25f, -0.125f, 0.f, 0.125f};
    const float SY[9] = {-0.125f, -0.25f, -0.125f, 0.f, 0.f, 0.f, 0.125f, 0.25f, 0.125f};
    const float LP[9] = {0.f, -1.f/6.f, 0.f, -1.f/6.f, 4.f/6.f, -1.f/6.f, 0.f, -1.f/6.f, 0.f};
    const float CS[9] = {-1.f/9.f, -1.f/9.f, -1.f/9.f, -1.f/9.f, 8.f/9.f, -1.f/9.f,
                         -1.f/9.f, -1.f/9.f, -1.f/9.f};
    float* kd = kcomb + (size_t)(b * 64 + t) * 12;
    #pragma unroll
    for (int k = 0; k < 9; ++k) {
        // feats order: [idg, center_surround, sobel_x, sobel_y, laplacian, vanilla]
        float v = att[0] * idg_w[t * 9 + k];
        v = fmaf(att[1], CS[k], v);
        v = fmaf(att[2], SX[k], v);
        v = fmaf(att[3], SY[k], v);
        v = fmaf(att[4], LP[k], v);
        v = fmaf(att[5], van_w[t * 9 + k], v);
        kd[k] = v;
    }
    kd[9] = 0.f; kd[10] = 0.f; kd[11] = 0.f;
}

// ---------------------------------------------------------------------------
// Kernel C: fused combined-dwconv (fp32) + 1x1 fusion conv via bf16 WMMA.
// grid = (W/128, H, B) = (2, 256, 8), block = 256 (8 waves).
// Each block: batch b, row h, 128 consecutive pixels, all 64 channels.
// ---------------------------------------------------------------------------
__global__ __launch_bounds__(256) void dgconv_fused_kernel(
    const float* __restrict__ x, const float* __restrict__ kcomb,
    const unsigned short* __restrict__ fusw_bf, const float* __restrict__ fus_b,
    float* __restrict__ out) {
    const int w0   = blockIdx.x * 128;
    const int h    = blockIdx.y;
    const int b    = blockIdx.z;
    const int tid  = threadIdx.x;
    const int wave = tid >> 5;
    const int lane = tid & 31;

    // tmp[p][c] in bf16, row stride 36 uints = 72 bf16 (144B, 16B-aligned rows)
    __shared__ unsigned sTmp[128 * 36];

    // ---- Step 1: depthwise conv with combined per-(b,c) 3x3 kernel --------
    const size_t xb = (size_t)b * 64u * 65536u;
    for (int it = 0; it < 16; ++it) {
        const int task = wave * 16 + it;     // 128 tasks: 32 ch-pairs x 4 px-groups
        const int cp = task >> 2;
        const int pg = task & 3;
        const int c0 = cp * 2;
        const int px = pg * 32 + lane;
        const int w  = w0 + px;
        const float* k0 = kcomb + (size_t)(b * 64 + c0) * 12;
        const float* k1 = k0 + 12;
        const float* x0 = x + xb + (size_t)c0 * 65536u;
        const float* x1 = x0 + 65536u;
        float a0 = 0.f, a1 = 0.f;
        #pragma unroll
        for (int dy = -1; dy <= 1; ++dy) {
            const int hh = h + dy;
            const bool rv = (hh >= 0) & (hh < 256);
            #pragma unroll
            for (int dx = -1; dx <= 1; ++dx) {
                const int ww = w + dx;
                const bool ok = rv & (ww >= 0) & (ww < 256);
                const int off = hh * 256 + ww;
                const float xv0 = ok ? x0[off] : 0.f;
                const float xv1 = ok ? x1[off] : 0.f;
                const int ki = (dy + 1) * 3 + (dx + 1);
                a0 = fmaf(k0[ki], xv0, a0);
                a1 = fmaf(k1[ki], xv1, a1);
            }
        }
        sTmp[px * 36 + cp] = (unsigned)f2bf(a0) | ((unsigned)f2bf(a1) << 16);
    }
    __syncthreads();

    // ---- Step 2: B fragments (tmp tile, N=pixel, K=channel) ----------------
    // B 32x16 as loaded here (same K order as A): lanes 0-15 -> K 0..15, lanes 16-31 -> K 16..31
    const int pcol = wave * 16 + (lane & 15);          // pixel = WMMA column n
    const int koff = (lane & 16) ? 16 : 0;
    const v8bf* trow = (const v8bf*)(sTmp + pcol * 36);
    const v8bf t00 = trow[(koff >> 3) + 0];            // c = koff .. koff+7
    const v8bf t01 = trow[(koff >> 3) + 1];            // c = koff+8 .. koff+15
    const v8bf t10 = trow[(koff >> 3) + 4];            // c = 32+koff ..
    const v8bf t11 = trow[(koff >> 3) + 5];
    const v16bf B0 = CAT16(t00, t01);                  // K-half 0: channels 0..31
    const v16bf B1 = CAT16(t10, t11);                  // K-half 1: channels 32..63

    // ---- Step 3: GEMM out[o,p] = sum_c fus_w[o,c] * tmp[p,c]  (+ bias) -----
    // ISA 16-bit A 16x32: lanes 0-15 -> K {0..7,16..23}, lanes 16-31 -> +8
    const int ohalf = (lane & 16) ? 8 : 0;
    const int kb    = (lane & 16) ? 8 : 0;
    float* const outp = out + (((size_t)b * 64u) * 65536u) + (size_t)h * 256u + w0
                        + wave * 16 + (lane & 15);
    #pragma unroll
    for (int mt = 0; mt < 4; ++mt) {
        const int orow = mt * 16 + (lane & 15);        // WMMA row m -> output ch
        const v8bf* wrow = (const v8bf*)(fusw_bf + orow * 64);
        const v8bf a00 = wrow[(2 * kb)      >> 3];     // c = 2kb .. 2kb+7
        const v8bf a01 = wrow[(8 + 2 * kb)  >> 3];     // c = 2kb+8 ..
        const v8bf a10 = wrow[(32 + 2 * kb) >> 3];
        const v8bf a11 = wrow[(40 + 2 * kb) >> 3];
        const v16bf A0 = CAT16(a00, a01);
        const v16bf A1 = CAT16(a10, a11);

        const int ob = mt * 16 + ohalf;                // D VGPR j -> o = ob + j
        v8f acc;
        #pragma unroll
        for (int j = 0; j < 8; ++j) acc[j] = fus_b[ob + j];

        acc = __builtin_amdgcn_wmma_f32_16x16x32_bf16(false, A0, false, B0,
                                                      (short)0, acc, false, false);
        acc = __builtin_amdgcn_wmma_f32_16x16x32_bf16(false, A1, false, B1,
                                                      (short)0, acc, false, false);

        #pragma unroll
        for (int j = 0; j < 8; ++j)                    // two 64B segments / store
            outp[(size_t)(ob + j) * 65536u] = acc[j];
    }
}

// ---------------------------------------------------------------------------
extern "C" void kernel_launch(void* const* d_in, const int* in_sizes, int n_in,
                              void* d_out, int out_size, void* d_ws, size_t ws_size,
                              hipStream_t stream) {
    (void)in_sizes; (void)n_in; (void)out_size; (void)ws_size;
    const float* x     = (const float*)d_in[0];
    const float* idg_w = (const float*)d_in[1];
    const float* van_w = (const float*)d_in[2];
    const float* w1    = (const float*)d_in[3];
    const float* b1    = (const float*)d_in[4];
    const float* w2    = (const float*)d_in[5];
    const float* b2    = (const float*)d_in[6];
    const float* fus_w = (const float*)d_in[7];
    const float* fus_b = (const float*)d_in[8];
    float* out = (float*)d_out;

    char* ws = (char*)d_ws;
    float*          pooled  = (float*)ws;                       // 512 f  (2 KB)
    float*          kcomb   = (float*)(ws + 4096);              // 8*64*12 f (24 KB)
    unsigned short* fusw_bf = (unsigned short*)(ws + 4096 + 32768); // 4096 bf16 (8 KB)

    dgconv_pool_kernel<<<512, 256, 0, stream>>>(x, fus_w, pooled, fusw_bf);
    dgconv_att_kernel<<<8, 64, 0, stream>>>(pooled, idg_w, van_w, w1, b1, w2, b2, kcomb);
    dgconv_fused_kernel<<<dim3(2, 256, 8), 256, 0, stream>>>(x, kcomb, fusw_bf, fus_b, out);
}
